// Mamba_3178275799301
// MI455X (gfx1250) — matrix operand order repaired and checked by reading, compiled, not run on hardware
//
#include <hip/hip_runtime.h>
#include <hip/hip_bf16.h>
#include <stdint.h>

#define ED   2048
#define DM   1024
#define NST  16
#define DTR  64
#define BB   2
#define LL   1024
#define NBL  (BB * LL)   // 2048 token rows
#define DBC  96          // dt_rank + 2*N

// GEMM tiling
#define BM   64
#define BN   256
#define BK   32
#define LDSROW 40        // 32 data elems + 4-DWORD pad (80B row stride)

typedef __bf16 bf16_t;
typedef __attribute__((ext_vector_type(8)))  bf16_t v8bf;
typedef __attribute__((ext_vector_type(16))) bf16_t v16bf;
typedef __attribute__((ext_vector_type(8)))  float  v8f;
typedef __attribute__((ext_vector_type(4)))  unsigned int u32x4;
typedef __attribute__((ext_vector_type(8)))  int i32x8;
typedef __attribute__((ext_vector_type(4)))  int i32x4;

#if defined(__has_builtin)
#if __has_builtin(__builtin_amdgcn_tensor_load_to_lds) && \
    __has_builtin(__builtin_amdgcn_s_wait_tensorcnt)
#define USE_TDM 1
#endif
#endif
#ifndef USE_TDM
#define USE_TDM 0
#endif

static __device__ __forceinline__ bf16_t f2bf(float f) {
  union { float f; uint32_t u; } c; c.f = f;
  uint32_t u = c.u;
  uint32_t r = (u + 0x7FFFu + ((u >> 16) & 1u)) >> 16;   // RNE
  union { uint16_t s; bf16_t b; } o; o.s = (uint16_t)r;
  return o.b;
}

static __device__ __forceinline__ float sigmoidf_(float x) {
  return 1.0f / (1.0f + __expf(-x));
}

static __device__ __forceinline__ v16bf load_afrag(const bf16_t* p) {
  v8bf a0 = *(const v8bf*)(p);
  v8bf a1 = *(const v8bf*)(p + 16);
  return __builtin_shufflevector(a0, a1,
      0,1,2,3,4,5,6,7,8,9,10,11,12,13,14,15);
}

static __device__ __forceinline__ uint32_t lds_of(const void* p) {
  // ISA 10.2: flat addr low 32 bits == LDS byte offset
  return (uint32_t)(uintptr_t)p;
}

#if USE_TDM
// Issue one TDM 2D tile load: tileRows x 32 (bf16) into LDS with padding
// (pad 4 DWORDs after every 16 -> LDS row stride 80B = LDSROW elems).
// D# packing per CDNA5 ISA ch.8.3/8.4. 6-arg builtin (groups 2/3 unused).
static __device__ __forceinline__ void tdm_load_tile(
    uint32_t ldsOff, const bf16_t* gtile, int tileRows,
    int rowsRemain, int colsRemain, int ld) {
  uint64_t ga = (uint64_t)(uintptr_t)gtile;
  u32x4 g0;
  g0[0] = 1u;                                             // count=1, user mode
  g0[1] = ldsOff;                                         // lds_addr
  g0[2] = (uint32_t)(ga & 0xFFFFFFFFu);                   // global_addr[31:0]
  g0[3] = (uint32_t)((ga >> 32) & 0x01FFFFFFu)            // global_addr[56:32]
        | 0x80000000u;                                    // type=2 (image)
  uint32_t td0 = (uint32_t)colsRemain;                    // tensor_dim0 (elems)
  uint32_t td1 = (uint32_t)rowsRemain;                    // tensor_dim1 (rows)
  uint64_t st0 = (uint64_t)(uint32_t)ld;                  // tensor_dim0_stride
  i32x8 g1;
  g1[0] = (int)((1u << 16)      // data_size = 2 bytes
              | (1u << 20)      // pad_enable
              | (3u << 22)      // pad_interval: every 16 DWORDs
              | (3u << 25));    // pad_amount: 4 DWORDs
  g1[1] = (int)((td0 & 0xFFFFu) << 16);                          // td0[15:0]
  g1[2] = (int)(((td0 >> 16) & 0xFFFFu) | ((td1 & 0xFFFFu) << 16));
  g1[3] = (int)(((td1 >> 16) & 0xFFFFu) | ((uint32_t)BK << 16)); // tile_dim0
  g1[4] = (int)((uint32_t)tileRows & 0xFFFFu);                   // tile_dim1
  g1[5] = (int)(uint32_t)(st0 & 0xFFFFFFFFu);
  g1[6] = (int)(uint32_t)((st0 >> 32) & 0xFFFFu);
  g1[7] = 0;
  i32x4 z4 = {0, 0, 0, 0};
  i32x8 z8 = {0, 0, 0, 0, 0, 0, 0, 0};
  __builtin_amdgcn_tensor_load_to_lds(g0, g1, z4, z4, z8, 0);
}
#endif

// ---------------------------------------------------------------- converters
__global__ __launch_bounds__(256) void k_cvt_bf16(
    const float* __restrict__ src, bf16_t* __restrict__ dst,
    int nrows, int rowlen, int sstride, int dstride) {
  int idx = blockIdx.x * blockDim.x + threadIdx.x;
  int total = nrows * rowlen;
  if (idx >= total) return;
  int r = idx / rowlen;
  int c = idx - r * rowlen;
  dst[(size_t)r * dstride + c] = f2bf(src[(size_t)r * sstride + c]);
}

// ---------------------------------------------------------------- WMMA GEMM
// C[m][n] = sum_k A[m][k] * W[n][k]   (A @ W^T), bf16 in, f32 out.
// Block = 256 threads = 8 waves arranged 2(M) x 4(N); block tile 64 x 256.
// A/B tiles staged in LDS by the Tensor Data Mover, double buffered;
// wave tile 32x64 -> 8 wmma per k-step, fragments read via ds_load_b128.
__global__ __launch_bounds__(256) void k_gemm_bf16(
    const bf16_t* __restrict__ A, const bf16_t* __restrict__ W,
    float* __restrict__ C, int M, int N, int K,
    int lda, int ldb, int ldc) {
  __shared__ bf16_t sA[2][BM * LDSROW];   // 2 x  5 KB
  __shared__ bf16_t sB[2][BN * LDSROW];   // 2 x 20 KB

  const int tid  = threadIdx.x;
  const int lane = tid & 31;
  const int wid  = tid >> 5;
  const int half = lane >> 4;
  const int l16  = lane & 15;

  const int mBlk = blockIdx.y * BM;
  const int nBlk = blockIdx.x * BN;
  const int mW   = (wid >> 2) * 32;       // wave M offset in tile
  const int nW   = (wid & 3) * 64;        // wave N offset in tile

  v8f acc00 = {}, acc01 = {}, acc02 = {}, acc03 = {};
  v8f acc10 = {}, acc11 = {}, acc12 = {}, acc13 = {};

  // LDS fragment addresses (row stride LDSROW elems)
  const int arow0 = mW + l16, arow1 = mW + 16 + l16;
  const int bc0 = nW      + l16;
  const int bc1 = nW + 16 + l16;
  const int bc2 = nW + 32 + l16;
  const int bc3 = nW + 48 + l16;

#if USE_TDM
  auto issue = [&](int k, int buf) {
    if (wid == 0) {
      tdm_load_tile(lds_of(&sA[buf][0]), A + (size_t)mBlk * lda + k,
                    BM, M - mBlk, K - k, lda);
      tdm_load_tile(lds_of(&sB[buf][0]), W + (size_t)nBlk * ldb + k,
                    BN, N - nBlk, K - k, ldb);
    }
  };
  issue(0, 0);
  if (wid == 0) __builtin_amdgcn_s_wait_tensorcnt((short)0);
  __syncthreads();
#else
  auto coop = [&](int k, int buf) {
    // cooperative fallback copy with OOB clamping
    for (int i = tid; i < BM * 2; i += 256) {       // A: 2 chunks of 16 elems/row
      int r = i >> 1, h = i & 1;
      int gr = mBlk + r; if (gr >= M) gr = M - 1;
      *(v8bf*)&sA[buf][r * LDSROW + h * 16] =
          *(const v8bf*)(A + (size_t)gr * lda + k + h * 16);
      *(v8bf*)&sA[buf][r * LDSROW + h * 16 + 8] =
          *(const v8bf*)(A + (size_t)gr * lda + k + h * 16 + 8);
    }
    for (int i = tid; i < BN * 2; i += 256) {
      int r = i >> 1, h = i & 1;
      int gr = nBlk + r; if (gr >= N) gr = N - 1;
      *(v8bf*)&sB[buf][r * LDSROW + h * 16] =
          *(const v8bf*)(W + (size_t)gr * ldb + k + h * 16);
      *(v8bf*)&sB[buf][r * LDSROW + h * 16 + 8] =
          *(const v8bf*)(W + (size_t)gr * ldb + k + h * 16 + 8);
    }
  };
  coop(0, 0);
  __syncthreads();
#endif

  int cur = 0;
  for (int k = 0; k < K; k += BK) {
    const bool more = (k + BK) < K;
#if USE_TDM
    if (more) issue(k + BK, cur ^ 1);     // DMA next tile while computing
#endif
    // ---- fragments from LDS
    v16bf af0 = load_afrag(&sA[cur][arow0 * LDSROW + half * 8]);
    v16bf af1 = load_afrag(&sA[cur][arow1 * LDSROW + half * 8]);
    v16bf b0 = load_afrag(&sB[cur][bc0 * LDSROW + half * 8]);
    v16bf b1 = load_afrag(&sB[cur][bc1 * LDSROW + half * 8]);
    v16bf b2 = load_afrag(&sB[cur][bc2 * LDSROW + half * 8]);
    v16bf b3 = load_afrag(&sB[cur][bc3 * LDSROW + half * 8]);

    acc00 = __builtin_amdgcn_wmma_f32_16x16x32_bf16(false, af0, false, b0,
                                                    (short)0, acc00, false, false);
    acc01 = __builtin_amdgcn_wmma_f32_16x16x32_bf16(false, af0, false, b1,
                                                    (short)0, acc01, false, false);
    acc02 = __builtin_amdgcn_wmma_f32_16x16x32_bf16(false, af0, false, b2,
                                                    (short)0, acc02, false, false);
    acc03 = __builtin_amdgcn_wmma_f32_16x16x32_bf16(false, af0, false, b3,
                                                    (short)0, acc03, false, false);
    acc10 = __builtin_amdgcn_wmma_f32_16x16x32_bf16(false, af1, false, b0,
                                                    (short)0, acc10, false, false);
    acc11 = __builtin_amdgcn_wmma_f32_16x16x32_bf16(false, af1, false, b1,
                                                    (short)0, acc11, false, false);
    acc12 = __builtin_amdgcn_wmma_f32_16x16x32_bf16(false, af1, false, b2,
                                                    (short)0, acc12, false, false);
    acc13 = __builtin_amdgcn_wmma_f32_16x16x32_bf16(false, af1, false, b3,
                                                    (short)0, acc13, false, false);

    if (more) {
#if USE_TDM
      if (wid == 0) __builtin_amdgcn_s_wait_tensorcnt((short)0);
      __syncthreads();                    // publish next buffer
#else
      __syncthreads();
      coop(k + BK, cur ^ 1);
      __syncthreads();
#endif
      cur ^= 1;
    }
  }

  // ---- store D: element (v + 8*half, l16) per VGPR v   (ISA C/D layout)
  const int n0 = nBlk + nW      + l16;
  const int n1 = nBlk + nW + 16 + l16;
  const int n2 = nBlk + nW + 32 + l16;
  const int n3 = nBlk + nW + 48 + l16;
#pragma unroll
  for (int v = 0; v < 8; ++v) {
    int r0 = mBlk + mW + half * 8 + v;
    int r1 = r0 + 16;
    if (r0 < M) {
      float* crow = C + (size_t)r0 * ldc;
      if (n0 < N) crow[n0] = acc00[v];
      if (n1 < N) crow[n1] = acc01[v];
      if (n2 < N) crow[n2] = acc02[v];
      if (n3 < N) crow[n3] = acc03[v];
    }
    if (r1 < M) {
      float* crow = C + (size_t)r1 * ldc;
      if (n0 < N) crow[n0] = acc10[v];
      if (n1 < N) crow[n1] = acc11[v];
      if (n2 < N) crow[n2] = acc12[v];
      if (n3 < N) crow[n3] = acc13[v];
    }
  }
}

// ------------------------------------------- causal depthwise conv4 + SiLU
__global__ __launch_bounds__(256) void k_conv_silu(
    const float* __restrict__ xz, const float* __restrict__ conv_w,
    float* __restrict__ xs, bf16_t* __restrict__ xs_bf) {
  int gid = blockIdx.x * blockDim.x + threadIdx.x;
  if (gid >= NBL * ED) return;
  int e  = gid & (ED - 1);
  int bl = gid >> 11;
  int l  = bl & (LL - 1);
  int b  = bl >> 10;

  float w0 = conv_w[e * 4 + 0];
  float w1 = conv_w[e * 4 + 1];
  float w2 = conv_w[e * 4 + 2];
  float w3 = conv_w[e * 4 + 3];

  const float* base = xz + ((size_t)b * LL) * (2 * ED) + e;
  float acc = w3 * base[(size_t)l * (2 * ED)];
  if (l >= 1) acc += w2 * base[(size_t)(l - 1) * (2 * ED)];
  if (l >= 2) acc += w1 * base[(size_t)(l - 2) * (2 * ED)];
  if (l >= 3) acc += w0 * base[(size_t)(l - 3) * (2 * ED)];

  float s = acc * sigmoidf_(acc);
  xs[gid]    = s;
  xs_bf[gid] = f2bf(s);
}

// ------------------------------------------------- delta bias + softplus
__global__ __launch_bounds__(256) void k_softplus_bias(
    float* __restrict__ delta, const float* __restrict__ dt_b) {
  int gid = blockIdx.x * blockDim.x + threadIdx.x;
  if (gid >= NBL * ED) return;
  int e = gid & (ED - 1);
  float v = delta[gid] + dt_b[e];
  delta[gid] = (v > 20.0f) ? v : log1pf(__expf(v));
}

// ------------------------------------------------- selective scan (wave32)
__global__ __launch_bounds__(256) void k_scan(
    const float* __restrict__ delta, const float* __restrict__ xs,
    const float* __restrict__ dBC, const float* __restrict__ A_log,
    const float* __restrict__ Dp, float* __restrict__ y) {
  int gtid = blockIdx.x * blockDim.x + threadIdx.x;
  int wave = gtid >> 5;
  int lane = gtid & 31;
  if (wave >= BB * (ED / 2)) return;

  int b  = wave / (ED / 2);
  int ep = wave - b * (ED / 2);
  int e  = ep * 2 + (lane >> 4);
  int n  = lane & 15;

  float negA = -__expf(A_log[(size_t)e * NST + n]);
  float Dv   = Dp[e];
  float h    = 0.0f;

  const size_t rowBL0 = (size_t)b * LL;
  for (int l = 0; l < LL; ++l) {
    size_t row = rowBL0 + l;
    float d  = delta[row * ED + e];
    float xv = xs[row * ED + e];
    const float* pbc = dBC + row * DBC;
    float Bv = pbc[DTR + n];
    float Cv = pbc[DTR + NST + n];

    h = __expf(d * negA) * h + (d * xv) * Bv;

    float yn = h * Cv;
    yn += __shfl_xor(yn, 8, 16);
    yn += __shfl_xor(yn, 4, 16);
    yn += __shfl_xor(yn, 2, 16);
    yn += __shfl_xor(yn, 1, 16);
    if (n == 0) y[row * ED + e] = yn + Dv * xv;
  }
}

// ------------------------------------------------- y * silu(z) -> bf16
__global__ __launch_bounds__(256) void k_gate(
    const float* __restrict__ y, const float* __restrict__ xz,
    bf16_t* __restrict__ yz_bf) {
  int gid = blockIdx.x * blockDim.x + threadIdx.x;
  if (gid >= NBL * ED) return;
  int e  = gid & (ED - 1);
  int bl = gid >> 11;
  float z = xz[(size_t)bl * (2 * ED) + ED + e];
  float g = z * sigmoidf_(z);
  yz_bf[gid] = f2bf(y[gid] * g);
}

// =========================================================================
extern "C" void kernel_launch(void* const* d_in, const int* in_sizes, int n_in,
                              void* d_out, int out_size, void* d_ws, size_t ws_size,
                              hipStream_t stream) {
  (void)in_sizes; (void)n_in; (void)out_size; (void)ws_size;
  const float* x        = (const float*)d_in[0];
  const float* in_w     = (const float*)d_in[1];
  const float* conv_w   = (const float*)d_in[2];
  const float* xproj_w  = (const float*)d_in[3];
  const float* dt_w     = (const float*)d_in[4];
  const float* dt_b     = (const float*)d_in[5];
  const float* A_log    = (const float*)d_in[6];
  const float* Dp       = (const float*)d_in[7];
  const float* out_w    = (const float*)d_in[8];
  float* out = (float*)d_out;

  uint8_t* ws = (uint8_t*)d_ws;
  size_t off = 0;
  auto carve = [&](size_t bytes) -> uint8_t* {
    uint8_t* p = ws + off;
    off += (bytes + 255) & ~(size_t)255;
    return p;
  };
  bf16_t* x_bf     = (bf16_t*)carve((size_t)NBL * DM * 2);
  bf16_t* w_in_bf  = (bf16_t*)carve((size_t)2 * ED * DM * 2);
  bf16_t* w_xp_bf  = (bf16_t*)carve((size_t)DBC * ED * 2);
  bf16_t* w_dt_bf  = (bf16_t*)carve((size_t)ED * DTR * 2);
  bf16_t* w_out_bf = (bf16_t*)carve((size_t)DM * ED * 2);
  float*  xz       = (float*) carve((size_t)NBL * 2 * ED * 4);
  float*  xs       = (float*) carve((size_t)NBL * ED * 4);
  bf16_t* xs_bf    = (bf16_t*)carve((size_t)NBL * ED * 2);
  float*  dBC      = (float*) carve((size_t)NBL * DBC * 4);
  bf16_t* dp_bf    = (bf16_t*)carve((size_t)NBL * DTR * 2);
  float*  delta    = (float*) carve((size_t)NBL * ED * 4);
  float*  ybuf     = (float*) carve((size_t)NBL * ED * 4);
  bf16_t* yz_bf    = (bf16_t*)carve((size_t)NBL * ED * 2);

  const int TPB = 256;
  auto blocks = [&](size_t n) { return (int)((n + TPB - 1) / TPB); };

  k_cvt_bf16<<<blocks((size_t)NBL * DM), TPB, 0, stream>>>(x, x_bf, NBL, DM, DM, DM);
  k_cvt_bf16<<<blocks((size_t)2 * ED * DM), TPB, 0, stream>>>(in_w, w_in_bf, 2 * ED, DM, DM, DM);
  k_cvt_bf16<<<blocks((size_t)DBC * ED), TPB, 0, stream>>>(xproj_w, w_xp_bf, DBC, ED, ED, ED);
  k_cvt_bf16<<<blocks((size_t)ED * DTR), TPB, 0, stream>>>(dt_w, w_dt_bf, ED, DTR, DTR, DTR);
  k_cvt_bf16<<<blocks((size_t)DM * ED), TPB, 0, stream>>>(out_w, w_out_bf, DM, ED, ED, ED);

  // GEMM 1: xz = x @ in_proj_w^T
  {
    dim3 grid((2 * ED) / BN, NBL / BM);
    k_gemm_bf16<<<grid, TPB, 0, stream>>>(x_bf, w_in_bf, xz,
                                          NBL, 2 * ED, DM, DM, DM, 2 * ED);
  }

  k_conv_silu<<<blocks((size_t)NBL * ED), TPB, 0, stream>>>(xz, conv_w, xs, xs_bf);

  // GEMM 2: dBC = xs @ x_proj_w^T   (N=96 handled by TDM OOB zero-fill + store guards)
  {
    dim3 grid(1, NBL / BM);
    k_gemm_bf16<<<grid, TPB, 0, stream>>>(xs_bf, w_xp_bf, dBC,
                                          NBL, DBC, ED, ED, ED, DBC);
  }

  k_cvt_bf16<<<blocks((size_t)NBL * DTR), TPB, 0, stream>>>(dBC, dp_bf, NBL, DTR, DBC, DTR);

  // GEMM 3: deltaRaw = dpart @ dt_w^T
  {
    dim3 grid(ED / BN, NBL / BM);
    k_gemm_bf16<<<grid, TPB, 0, stream>>>(dp_bf, w_dt_bf, delta,
                                          NBL, ED, DTR, DTR, DTR, ED);
  }

  k_softplus_bias<<<blocks((size_t)NBL * ED), TPB, 0, stream>>>(delta, dt_b);

  {
    int waves = BB * (ED / 2);
    k_scan<<<blocks((size_t)waves * 32), TPB, 0, stream>>>(delta, xs, dBC,
                                                           A_log, Dp, ybuf);
  }

  k_gate<<<blocks((size_t)NBL * ED), TPB, 0, stream>>>(ybuf, xz, yz_bf);

  // GEMM 4: out = yz @ out_proj_w^T
  {
    dim3 grid(DM / BN, NBL / BM);
    k_gemm_bf16<<<grid, TPB, 0, stream>>>(yz_bf, w_out_bf, out,
                                          NBL, DM, ED, ED, ED, DM);
  }
}
